// PhaseAwareQuantization_72602127172073
// MI455X (gfx1250) — matrix operand order repaired and checked by reading, not compiled, not run on hardware
//
#include <hip/hip_runtime.h>

typedef __attribute__((ext_vector_type(16))) _Float16 v16h;
typedef __attribute__((ext_vector_type(8)))  float    v8f;

constexpr int  B  = 32;
constexpr int  C  = 9;
constexpr int  T  = 4096;          // power of two, log2 = 12
constexpr int  BT = B * T;         // 131072 rows
constexpr int  NC = 512;           // codebook entries
constexpr int  D  = 64;            // code dim

// ---------------------------------------------------------------------------
// Kernel 1: zero the phase accumulator
// ---------------------------------------------------------------------------
__global__ void zero_f32(float* __restrict__ p, int n) {
  int i = blockIdx.x * blockDim.x + threadIdx.x;
  if (i < n) p[i] = 0.f;
}

// ---------------------------------------------------------------------------
// Kernel 2: analytic-signal phase via in-LDS FFT (one (b,c) signal per WG).
// forward FFT -> Hilbert multiplier -> inverse FFT -> atan2 -> atomic accum.
// 4096 complex f32 = 32KB LDS (WGP has 320KB). 256 threads = 8 wave32.
// ---------------------------------------------------------------------------
__global__ __launch_bounds__(256) void hilbert_phase(const float* __restrict__ imu,
                                                     float* __restrict__ phase_sum) {
  __shared__ float re[T];
  __shared__ float im[T];
  const int bc  = blockIdx.x;            // b*C + c
  const int b   = bc / C;
  const int tid = threadIdx.x;
  const float* x = imu + (size_t)bc * T;

  // load with 12-bit bit-reversal for in-place DIT
  for (int i = tid; i < T; i += 256) {
    int r = (int)(__brev((unsigned)i) >> 20);
    re[r] = x[i];
    im[r] = 0.f;
  }
  __syncthreads();

  // forward FFT (DIT): w = exp(-2*pi*i*k/len)
  for (int s = 0; s < 12; ++s) {
    const int   half = 1 << s;
    const float ang  = -6.28318530717958647692f / (float)(2 << s);
    for (int j = tid; j < (T / 2); j += 256) {
      const int k    = j & (half - 1);
      const int base = (j >> s) << (s + 1);
      float wi, wr;
      __sincosf(ang * (float)k, &wi, &wr);
      const int i0 = base + k, i1 = i0 + half;
      float ar = re[i0], ai = im[i0];
      float br = re[i1], bi = im[i1];
      float tr = br * wr - bi * wi;
      float ti = br * wi + bi * wr;
      re[i0] = ar + tr; im[i0] = ai + ti;
      re[i1] = ar - tr; im[i1] = ai - ti;
    }
    __syncthreads();
  }

  // Hilbert multiplier in natural order: h[0]=1, h[1..T/2-1]=2, h[T/2]=1, rest 0
  for (int i = tid; i < T; i += 256) {
    float h = (i == 0 || i == T / 2) ? 1.f : (i < T / 2 ? 2.f : 0.f);
    re[i] *= h; im[i] *= h;
  }
  __syncthreads();

  // bit-reverse permute (pairwise swap) for inverse DIT
  for (int i = tid; i < T; i += 256) {
    int r = (int)(__brev((unsigned)i) >> 20);
    if (i < r) {
      float t0 = re[i]; re[i] = re[r]; re[r] = t0;
      float t1 = im[i]; im[i] = im[r]; im[r] = t1;
    }
  }
  __syncthreads();

  // inverse FFT; the 1/N scale is skipped (atan2 is invariant to positive scale)
  for (int s = 0; s < 12; ++s) {
    const int   half = 1 << s;
    const float ang  = 6.28318530717958647692f / (float)(2 << s);
    for (int j = tid; j < (T / 2); j += 256) {
      const int k    = j & (half - 1);
      const int base = (j >> s) << (s + 1);
      float wi, wr;
      __sincosf(ang * (float)k, &wi, &wr);
      const int i0 = base + k, i1 = i0 + half;
      float ar = re[i0], ai = im[i0];
      float br = re[i1], bi = im[i1];
      float tr = br * wr - bi * wi;
      float ti = br * wi + bi * wr;
      re[i0] = ar + tr; im[i0] = ai + ti;
      re[i1] = ar - tr; im[i1] = ai - ti;
    }
    __syncthreads();
  }

  // accumulate per-channel phase into (b, t) sum
  for (int i = tid; i < T; i += 256)
    atomicAdd(&phase_sum[(size_t)b * T + i], atan2f(im[i], re[i]));
}

// ---------------------------------------------------------------------------
// Kernel 3: codebook -> f16 copy + 0.5*||c||^2 (argmin d2 == argmax f.c-0.5||c||^2)
// ---------------------------------------------------------------------------
__global__ void codebook_prep(const float* __restrict__ cb,
                              _Float16* __restrict__ cb16,
                              float* __restrict__ cnorm_half) {
  int n = blockIdx.x * blockDim.x + threadIdx.x;
  if (n >= NC) return;
  float s = 0.f;
  for (int k = 0; k < D; ++k) {
    float v = cb[n * D + k];
    s += v * v;
    cb16[n * D + k] = (_Float16)v;
  }
  cnorm_half[n] = 0.5f * s;
}

// ---------------------------------------------------------------------------
// Kernel 4: per-(b,t) feature row: two 9->32 projections (f32 math), write
// f16 feature matrix for WMMA + the mean-phase output.
// ---------------------------------------------------------------------------
__global__ __launch_bounds__(256) void features_kernel(
    const float* __restrict__ imu, const float* __restrict__ phase_sum,
    const float* __restrict__ Wm, const float* __restrict__ bm,
    const float* __restrict__ Wp, const float* __restrict__ bp,
    _Float16* __restrict__ featA, float* __restrict__ phases_out) {
  __shared__ float sWm[32 * 9], sWp[32 * 9], sbm[32], sbp[32];
  const int tid = threadIdx.x;
  for (int i = tid; i < 32 * 9; i += 256) { sWm[i] = Wm[i]; sWp[i] = Wp[i]; }
  if (tid < 32) { sbm[tid] = bm[tid]; sbp[tid] = bp[tid]; }
  __syncthreads();

  const int row = blockIdx.x * 256 + tid;      // exact grid: row < BT
  const int b = row / T, t = row % T;

  float v[9];
#pragma unroll
  for (int c = 0; c < C; ++c) v[c] = imu[((size_t)b * C + c) * T + t];

  const float ph = phase_sum[row] * (1.f / 9.f);
  phases_out[row] = ph;
  float sp, cp;
  __sincosf(ph, &sp, &cp);

  _Float16* out = featA + (size_t)row * D;
#pragma unroll 4
  for (int o = 0; o < 32; ++o) {
    float a = sbm[o];
#pragma unroll
    for (int c = 0; c < 9; ++c) a += sWm[o * 9 + c] * v[c];
    out[o] = (_Float16)a;
  }
  float u[9] = {v[0], v[1], v[2], v[3], v[4], v[5], v[6], cp, sp};
#pragma unroll 4
  for (int o = 0; o < 32; ++o) {
    float a = sbp[o];
#pragma unroll
    for (int c = 0; c < 9; ++c) a += sWp[o * 9 + c] * u[c];
    out[32 + o] = (_Float16)a;
  }
}

// ---------------------------------------------------------------------------
// Kernel 5: fused GEMM + argmax via v_wmma_f32_16x16x32_f16.
// Block = 8 wave32, each wave owns TWO 16-row tiles (256 rows/block).
// The codebook is staged through LDS in two 32KB passes of 256 codes so the
// 8 waves share one global fetch; B fragments come from ds_load_b128.
// Per 16-column chunk: 4 wmma (2 tiles x 2 k-chunks) feed a running
// per-lane argmax of (f.c - 0.5||c||^2); final 16-lane shfl_xor reduction
// matches the 16x16 f32 C/D layout (lane0-15 -> M=vgpr, lane16-31 -> M=vgpr+8).
// ---------------------------------------------------------------------------
__global__ __launch_bounds__(256) void nn_argmax_wmma(
    const _Float16* __restrict__ featA, const _Float16* __restrict__ cb16,
    const float* __restrict__ cnorm_half, int* __restrict__ best_out) {
  constexpr int PASS_N = 256;                 // codes staged per pass (32KB)
  __shared__ _Float16 scb[PASS_N * D];
  __shared__ float    scn[PASS_N];

  const int lane  = threadIdx.x & 31;
  const int wave  = threadIdx.x >> 5;
  const int half  = lane >> 4;
  const int tile0 = (blockIdx.x * 8 + wave) * 2;   // two 16-row tiles per wave
  const int m0    = tile0 * 16 + (lane & 15);
  const int m1    = m0 + 16;

  // A 16x32 f16 layout (ISA 7.12.2): lane m = lane&15;
  // VGPR j: K = (j<4 ? 2j : 8+2j) + 8*half   (two contiguous halves per VGPR)
  v16h a00, a01, a10, a11;
  {
    const _Float16* f0 = featA + (size_t)m0 * D;
    const _Float16* f1 = featA + (size_t)m1 * D;
#pragma unroll
    for (int j = 0; j < 8; ++j) {
      const int k0 = ((j < 4) ? (2 * j) : (8 + 2 * j)) + half * 8;
      a00[2 * j] = f0[k0];      a00[2 * j + 1] = f0[k0 + 1];
      a01[2 * j] = f0[32 + k0]; a01[2 * j + 1] = f0[32 + k0 + 1];
      a10[2 * j] = f1[k0];      a10[2 * j + 1] = f1[k0 + 1];
      a11[2 * j] = f1[32 + k0]; a11[2 * j + 1] = f1[32 + k0 + 1];
    }
  }

  float best[2][8];
  int   bidx[2][8];
#pragma unroll
  for (int t = 0; t < 2; ++t)
#pragma unroll
    for (int i = 0; i < 8; ++i) { best[t][i] = -3.0e38f; bidx[t][i] = 0; }

  for (int pass = 0; pass < NC / PASS_N; ++pass) {
    const int nbase = pass * PASS_N;
    __syncthreads();   // previous-pass LDS readers done
    {
      const uint4* src = (const uint4*)(cb16 + (size_t)nbase * D);
      uint4* dst = (uint4*)scb;
      for (int i = threadIdx.x; i < (PASS_N * D) / 8; i += 256) dst[i] = src[i];
      if (threadIdx.x < PASS_N) scn[threadIdx.x] = cnorm_half[nbase + threadIdx.x];
    }
    __syncthreads();

#pragma unroll 2
    for (int chunk = 0; chunk < PASS_N / 16; ++chunk) {
      const int nl = chunk * 16 + (lane & 15);   // local code index in LDS
      const _Float16* cr = scb + nl * D;
      // B 32x16 f16: same K order as A: VGPR j: K = (j<4 ? 2j : 8+2j) + 8*half
      v16h b0, b1;
#pragma unroll
      for (int j = 0; j < 8; ++j) {
        const int k0 = ((j < 4) ? (2 * j) : (8 + 2 * j)) + half * 8;
        b0[2 * j] = cr[k0];      b0[2 * j + 1] = cr[k0 + 1];
        b1[2 * j] = cr[32 + k0]; b1[2 * j + 1] = cr[32 + k0 + 1];
      }
      v8f acc0 = {}, acc1 = {};
      acc0 = __builtin_amdgcn_wmma_f32_16x16x32_f16(false, a00, false, b0,
                                                    (short)0, acc0, false, false);
      acc1 = __builtin_amdgcn_wmma_f32_16x16x32_f16(false, a10, false, b0,
                                                    (short)0, acc1, false, false);
      acc0 = __builtin_amdgcn_wmma_f32_16x16x32_f16(false, a01, false, b1,
                                                    (short)0, acc0, false, false);
      acc1 = __builtin_amdgcn_wmma_f32_16x16x32_f16(false, a11, false, b1,
                                                    (short)0, acc1, false, false);
      const float cn = scn[nl];
      const int   n  = nbase + nl;
#pragma unroll
      for (int i = 0; i < 8; ++i) {
        const float s0 = acc0[i] - cn;          // f.c - 0.5||c||^2
        if (s0 > best[0][i]) { best[0][i] = s0; bidx[0][i] = n; }
        const float s1 = acc1[i] - cn;
        if (s1 > best[1][i]) { best[1][i] = s1; bidx[1][i] = n; }
      }
    }
  }

  // argmax across the 16 lanes holding different n (stays inside each half)
#pragma unroll
  for (int off = 1; off < 16; off <<= 1) {
#pragma unroll
    for (int t = 0; t < 2; ++t)
#pragma unroll
      for (int i = 0; i < 8; ++i) {
        const float ob = __shfl_xor(best[t][i], off, 32);
        const int   oi = __shfl_xor(bidx[t][i], off, 32);
        if (ob > best[t][i] || (ob == best[t][i] && oi < bidx[t][i])) {
          best[t][i] = ob; bidx[t][i] = oi;     // first-min tiebreak
        }
      }
  }

  if ((lane & 15) == 0) {
#pragma unroll
    for (int t = 0; t < 2; ++t) {
      const int base = (tile0 + t) * 16 + half * 8;  // lane0 rows 0..7, lane16 rows 8..15
#pragma unroll
      for (int i = 0; i < 8; ++i) best_out[base + i] = bidx[t][i];
    }
  }
}

// ---------------------------------------------------------------------------
// Kernel 6: gather f32 codebook rows into d_out + indices (as float)
// ---------------------------------------------------------------------------
__global__ void gather_out(const int* __restrict__ idx, const float* __restrict__ cb,
                           float* __restrict__ quant, float* __restrict__ idx_out) {
  const int gid = blockIdx.x * blockDim.x + threadIdx.x;   // BT*64 threads, exact
  const int row = gid >> 6;
  const int k   = gid & 63;
  const int id  = idx[row];
  quant[gid] = cb[id * D + k];
  if (k == 0) idx_out[row] = (float)id;
}

// ---------------------------------------------------------------------------
extern "C" void kernel_launch(void* const* d_in, const int* in_sizes, int n_in,
                              void* d_out, int out_size, void* d_ws, size_t ws_size,
                              hipStream_t stream) {
  const float* imu = (const float*)d_in[0];   // (B, 9, T)
  const float* Wm  = (const float*)d_in[1];   // (32, 9)
  const float* bm  = (const float*)d_in[2];   // (32,)
  const float* Wp  = (const float*)d_in[3];   // (32, 9)
  const float* bp  = (const float*)d_in[4];   // (32,)
  const float* cb  = (const float*)d_in[5];   // (512, 64)

  float* out   = (float*)d_out;
  float* quant = out;                         // BT*64
  float* idxo  = out + (size_t)BT * D;        // BT
  float* phout = idxo + BT;                   // BT

  char* w = (char*)d_ws;
  float*    phase_sum = (float*)(w);                                 //  512 KB
  int*      best_idx  = (int*)(w + 512u * 1024u);                    //  512 KB
  float*    cnorm     = (float*)(w + 1024u * 1024u);                 //    2 KB
  _Float16* cb16      = (_Float16*)(w + 1024u * 1024u + 64u * 1024u);//   64 KB
  _Float16* featA     = (_Float16*)(w + 1280u * 1024u);              //   16 MB

  zero_f32<<<(BT + 255) / 256, 256, 0, stream>>>(phase_sum, BT);
  hilbert_phase<<<B * C, 256, 0, stream>>>(imu, phase_sum);
  codebook_prep<<<(NC + 255) / 256, 256, 0, stream>>>(cb, cb16, cnorm);
  features_kernel<<<BT / 256, 256, 0, stream>>>(imu, phase_sum, Wm, bm, Wp, bp,
                                                featA, phout);
  nn_argmax_wmma<<<BT / 256, 256, 0, stream>>>(featA, cb16, cnorm, best_idx);
  gather_out<<<(BT * D) / 256, 256, 0, stream>>>(best_idx, cb, quant, idxo);
}
